// VQVAE_51788715655545
// MI455X (gfx1250) — matrix operand order repaired and checked by reading, not compiled, not run on hardware
//
#include <hip/hip_runtime.h>

// ---------------------------------------------------------------------------
// Types for CDNA5 WMMA (wave32, 16x16x32 bf16 -> f32)
// ---------------------------------------------------------------------------
typedef __bf16 bf16_t;
typedef bf16_t v16bf __attribute__((ext_vector_type(16)));
typedef bf16_t v8bf  __attribute__((ext_vector_type(8)));
typedef float  v8f   __attribute__((ext_vector_type(8)));

union FragU { v16bf v; v8bf h[2]; };

// Problem dims
constexpr int VN = 39;
constexpr int EN = 512;
constexpr int HN = 256;
constexpr int BN = 512;
constexpr int TN = 128;

// ---------------------------------------------------------------------------
// WMMA fragment loaders (ISA 7.12.2 layouts).
// A 16x32 bf16 (row-major, lane m = lane&15):
//   element e -> K = (e<8 ? e : e+8) + 8*hi  => two contiguous 8-elem chunks
//   at k0+8*hi and k0+16+8*hi.
// B 32x16 bf16 (stored as W[n][k] row-major; gemm B[k][n]=W[n][k]):
//   element e -> K = (e<8 ? e : e+8) + 8*hi (same as A).
// ---------------------------------------------------------------------------
__device__ inline v16bf load_fragA(const bf16_t* base, int m0, int stride, int k0) {
    int lane = threadIdx.x & 31;
    int hi = lane >> 4;
    const bf16_t* p = base + (m0 + (lane & 15)) * stride + k0 + 8 * hi;
    FragU f;
    f.h[0] = *(const v8bf*)(p);
    f.h[1] = *(const v8bf*)(p + 16);
    return f.v;
}

__device__ inline v16bf load_fragB(const bf16_t* base, int n0, int stride, int k0) {
    int lane = threadIdx.x & 31;
    int hi = lane >> 4;
    const bf16_t* p = base + (n0 + (lane & 15)) * stride + k0 + 8 * hi;
    FragU f;
    f.h[0] = *(const v8bf*)(p);
    f.h[1] = *(const v8bf*)(p + 16);
    return f.v;
}

// C = A(16 x K) * W(N x K)^T tile, f32 accumulate.
__device__ inline v8f wmma_tile(const bf16_t* A, int m0, int lda,
                                const bf16_t* W, int n0, int ldw,
                                int K, v8f acc) {
    for (int k0 = 0; k0 < K; k0 += 32) {
        v16bf a = load_fragA(A, m0, lda, k0);
        v16bf b = load_fragB(W, n0, ldw, k0);
        acc = __builtin_amdgcn_wmma_f32_16x16x32_bf16(false, a, false, b,
                                                      (short)0, acc, false, false);
    }
    return acc;
}

// D 16x16 f32 store: VGPR r holds M = r + 8*hi, N = lane&15.
__device__ inline void store_tile_f32(float* D, int m0, int ldd, int n0, v8f acc) {
    int lane = threadIdx.x & 31;
    int n = n0 + (lane & 15);
    int mb = m0 + ((lane >> 4) << 3);
    #pragma unroll
    for (int r = 0; r < 8; ++r)
        D[(mb + r) * ldd + n] = acc[r];
}

// ---------------------------------------------------------------------------
// Setup kernels
// ---------------------------------------------------------------------------
__global__ __launch_bounds__(256) void zero_kernel(float* p, int n) {
    for (int i = blockIdx.x * 256 + threadIdx.x; i < n; i += gridDim.x * 256)
        p[i] = 0.f;
}

// out[v][j] = sum_e W[j][coloff+e]*emb[v][e] + bias[j]   (E = 512)
__global__ __launch_bounds__(256) void table_kernel(
    const float* __restrict__ W, int wstride, int coloff,
    const float* __restrict__ bias,
    const float* __restrict__ emb, int rows, float* __restrict__ out) {
    int v = blockIdx.x;
    const float* e = emb + v * EN;
    for (int j = threadIdx.x; j < rows; j += 256) {
        float s = bias ? bias[j] : 0.f;
        const float* w = W + j * wstride + coloff;
        for (int k = 0; k < EN; ++k) s = fmaf(w[k], e[k], s);
        out[v * rows + j] = s;
    }
}

// dst (rows x cols bf16) = (bf16) src[r][coloff + c], src stride sstride
__global__ __launch_bounds__(256) void cvt_kernel(
    const float* __restrict__ src, int sstride, int coloff,
    bf16_t* __restrict__ dst, int rows, int cols) {
    int n = rows * cols;
    for (int i = blockIdx.x * 256 + threadIdx.x; i < n; i += gridDim.x * 256) {
        int r = i / cols, c = i - r * cols;
        dst[i] = (bf16_t)src[r * sstride + coloff + c];
    }
}

// ---------------------------------------------------------------------------
// Encoder step: both directions. grid = 64 blocks (32 batch chunks x 2 dirs),
// 256 threads (8 waves). gh = h @ Whh^T via WMMA, then fused GRU gates.
// ---------------------------------------------------------------------------
__global__ __launch_bounds__(256) void enc_step_kernel(
    int t,
    const int* __restrict__ tokens,       // B*T
    const float* __restrict__ giF,        // 39 x 768 (Wih_f@emb + bih_f)
    const float* __restrict__ giB,
    const bf16_t* __restrict__ WhhF,      // 768 x 256 bf16
    const bf16_t* __restrict__ WhhB,
    const float* __restrict__ bhhF,       // 768
    const float* __restrict__ bhhB,
    float* __restrict__ hF,               // 512 x 256
    float* __restrict__ hB,
    bf16_t* __restrict__ enc_bf) {        // (B*T) x 512
    __shared__ __align__(16) bf16_t s_hbf[16 * 256];
    __shared__ __align__(16) float  s_gh[16 * 768];

    int dir = blockIdx.x & 1;
    int row0 = (blockIdx.x >> 1) * 16;
    int pos = dir ? (TN - 1 - t) : t;
    const float*  gi_tab = dir ? giB : giF;
    const bf16_t* Whh    = dir ? WhhB : WhhF;
    const float*  bhh    = dir ? bhhB : bhhF;
    float*        h      = dir ? hB : hF;
    int tid = threadIdx.x;

    for (int i = tid; i < 16 * 256; i += 256)
        s_hbf[i] = (bf16_t)h[row0 * 256 + i];
    __syncthreads();

    int wave = tid >> 5;
    for (int it = 0; it < 6; ++it) {            // 48 N-tiles / 8 waves
        int nt = wave * 6 + it;
        v8f acc = {};
        acc = wmma_tile(s_hbf, 0, 256, Whh, nt * 16, 256, 256, acc);
        store_tile_f32(s_gh, 0, 768, nt * 16, acc);
    }
    __syncthreads();

    for (int i = tid; i < 16 * 256; i += 256) {
        int r = i >> 8, j = i & 255;
        int b = row0 + r;
        int tok = tokens[b * TN + pos];
        const float* gi = gi_tab + tok * 768;
        float ghr = s_gh[r * 768 + j]       + bhh[j];
        float ghz = s_gh[r * 768 + 256 + j] + bhh[256 + j];
        float ghn = s_gh[r * 768 + 512 + j] + bhh[512 + j];
        float rg = 1.f / (1.f + __expf(-(gi[j]       + ghr)));
        float zg = 1.f / (1.f + __expf(-(gi[256 + j] + ghz)));
        float ng = tanhf(gi[512 + j] + rg * ghn);
        float hp = h[b * 256 + j];
        float hn = (1.f - zg) * ng + zg * hp;
        h[b * 256 + j] = hn;
        enc_bf[(size_t)(b * TN + pos) * 512 + dir * 256 + j] = (bf16_t)hn;
    }
}

// hidden = tanh([hF,hB] @ bridge_W^T + b). grid 512, block 256.
__global__ __launch_bounds__(256) void bridge_kernel(
    const float* __restrict__ hF, const float* __restrict__ hB,
    const float* __restrict__ W, const float* __restrict__ bias,
    float* __restrict__ h_dec) {
    int b = blockIdx.x, i = threadIdx.x;
    float s = bias[i];
    const float* w = W + i * 512;
    for (int k = 0; k < 256; ++k) s = fmaf(w[k],       hF[b * 256 + k], s);
    for (int k = 0; k < 256; ++k) s = fmaf(w[256 + k], hB[b * 256 + k], s);
    h_dec[b * 256 + i] = tanhf(s);
}

// proj_key = enc_out @ Wk^T : (65536 x 512) x (512 x 256). grid 1024 x 256.
__global__ __launch_bounds__(256) void projkey_kernel(
    const bf16_t* __restrict__ enc_bf, const bf16_t* __restrict__ Wk,
    float* __restrict__ proj_key) {
    int wave_g = blockIdx.x * 8 + (threadIdx.x >> 5);
    for (int i = 0; i < 8; ++i) {               // 65536 tiles / 8192 waves
        int tile = wave_g * 8 + i;
        int mt = tile >> 4, nt = tile & 15;
        v8f acc = {};
        acc = wmma_tile(enc_bf, mt * 16, 512, Wk, nt * 16, 512, 512, acc);
        store_tile_f32(proj_key, mt * 16, 256, nt * 16, acc);
    }
}

// ---------------------------------------------------------------------------
// Decoder step: attention + GRU + pre + vocab loss, fully workgroup-local.
// grid 32 blocks x 16 batch rows, 256 threads, ~204KB dynamic LDS.
// ---------------------------------------------------------------------------
constexpr int SM_HBF  = 0;                       // 16*256 bf16  8192
constexpr int SM_WE   = 8192;                    // 256 f32      1024
constexpr int SM_Q    = 9216;                    // 16*256 f32   16384
constexpr int SM_ATT  = 25600;                   // 16*128 f32   8192
constexpr int SM_RST  = 33792;                   // 16*2 f32     128
constexpr int SM_CTXF = 33920;                   // 16*512 f32   32768
constexpr int SM_CTXB = 66688;                   // 16*512 bf16  16384
constexpr int SM_GH   = 83072;                   // 16*768 f32   49152
constexpr int SM_GI   = 132224;                  // 16*768 f32   49152
constexpr int SM_HNB  = 181376;                  // 16*256 bf16  8192
constexpr int SM_PRE  = 189568;                  // 16*256 f32   16384
constexpr int SM_LOG  = 205952;                  // 16*39 f32    2496
constexpr int SMEM_DEC = 208448;

__global__ __launch_bounds__(256) void dec_step_kernel(
    int t,
    const int* __restrict__ tokens,
    const float* __restrict__ proj_key,   // (B*T) x 256
    const bf16_t* __restrict__ enc_bf,    // (B*T) x 512
    const float* __restrict__ attn_We,    // 256
    const float* __restrict__ decE_tab,   // 39 x 768 (embed part of gi + bih)
    const float* __restrict__ preE_tab,   // 39 x 256 (embed part of pre)
    const bf16_t* __restrict__ Wq,        // 256 x 256
    const bf16_t* __restrict__ dWhh,      // 768 x 256
    const bf16_t* __restrict__ dWihC,     // 768 x 512 (ctx slice)
    const bf16_t* __restrict__ pWh,       // 256 x 256
    const bf16_t* __restrict__ pWc,       // 256 x 512
    const float* __restrict__ dec_bhh,    // 768
    const float* __restrict__ genW,       // 39 x 256
    float* __restrict__ h_dec,            // 512 x 256
    float* __restrict__ loss_part) {      // 127 x 512
    extern __shared__ __align__(16) char smem[];
    bf16_t* s_hbf  = (bf16_t*)(smem + SM_HBF);
    float*  s_we   = (float*) (smem + SM_WE);
    float*  s_q    = (float*) (smem + SM_Q);
    float*  s_att  = (float*) (smem + SM_ATT);
    float*  s_rst  = (float*) (smem + SM_RST);
    float*  s_ctx  = (float*) (smem + SM_CTXF);
    bf16_t* s_ctxb = (bf16_t*)(smem + SM_CTXB);
    float*  s_gh   = (float*) (smem + SM_GH);
    float*  s_gi   = (float*) (smem + SM_GI);
    bf16_t* s_hnb  = (bf16_t*)(smem + SM_HNB);
    float*  s_pre  = (float*) (smem + SM_PRE);
    float*  s_log  = (float*) (smem + SM_LOG);

    int tid = threadIdx.x;
    int row0 = blockIdx.x * 16;
    int wave = tid >> 5;

    // phase 1: stage h (bf16) and We
    for (int i = tid; i < 16 * 256; i += 256)
        s_hbf[i] = (bf16_t)h_dec[row0 * 256 + i];
    s_we[tid] = attn_We[tid];
    __syncthreads();

    // phase 2: q = h @ Wq^T (WMMA)
    for (int i = 0; i < 2; ++i) {
        int nt = wave * 2 + i;
        v8f acc = {};
        acc = wmma_tile(s_hbf, 0, 256, Wq, nt * 16, 256, 256, acc);
        store_tile_f32(s_q, 0, 256, nt * 16, acc);
    }
    __syncthreads();

    // phase 3: scores e[r][tt] = sum_k tanh(q+pk)*We, masked
    for (int i = tid; i < 16 * 128; i += 256) {
        int r = i >> 7, tt = i & 127;
        int b = row0 + r;
        const float* pk = proj_key + (size_t)(b * TN + tt) * 256;
        const float* q = s_q + r * 256;
        float s = 0.f;
        for (int k = 0; k < 256; ++k) s = fmaf(tanhf(q[k] + pk[k]), s_we[k], s);
        s_att[i] = (tokens[b * TN + tt] != 0) ? s : -3.4e38f;
    }
    __syncthreads();

    // phase 4: softmax over T
    if (tid < 16) {
        float m = -3.4e38f;
        for (int tt = 0; tt < 128; ++tt) m = fmaxf(m, s_att[tid * 128 + tt]);
        float sum = 0.f;
        for (int tt = 0; tt < 128; ++tt) sum += __expf(s_att[tid * 128 + tt] - m);
        s_rst[tid * 2] = m;
        s_rst[tid * 2 + 1] = sum;
    }
    __syncthreads();
    for (int i = tid; i < 16 * 128; i += 256) {
        int r = i >> 7;
        s_att[i] = __expf(s_att[i] - s_rst[r * 2]) / s_rst[r * 2 + 1];
    }
    __syncthreads();

    // phase 5: ctx[r][d] = sum_t a * enc_out
    for (int i = tid; i < 16 * 512; i += 256) {
        int r = i >> 9, d = i & 511;
        int b = row0 + r;
        const bf16_t* eo = enc_bf + (size_t)b * TN * 512 + d;
        const float* a = s_att + r * 128;
        float s = 0.f;
        for (int tt = 0; tt < 128; ++tt) s = fmaf(a[tt], (float)eo[tt * 512], s);
        s_ctx[i] = s;
        s_ctxb[i] = (bf16_t)s;
    }
    __syncthreads();

    // phase 6: gh = h @ dWhh^T (K=256), gi_ctx = ctx @ dWihC^T (K=512)
    for (int i = 0; i < 12; ++i) {              // 96 tiles / 8 waves
        int idx = wave * 12 + i;
        if (idx < 48) {
            v8f acc = {};
            acc = wmma_tile(s_hbf, 0, 256, dWhh, idx * 16, 256, 256, acc);
            store_tile_f32(s_gh, 0, 768, idx * 16, acc);
        } else {
            int nt = idx - 48;
            v8f acc = {};
            acc = wmma_tile(s_ctxb, 0, 512, dWihC, nt * 16, 512, 512, acc);
            store_tile_f32(s_gi, 0, 768, nt * 16, acc);
        }
    }
    __syncthreads();

    // phase 7: GRU gates
    for (int i = tid; i < 16 * 256; i += 256) {
        int r = i >> 8, j = i & 255;
        int b = row0 + r;
        int tok = tokens[b * TN + t];
        const float* ge = decE_tab + tok * 768;
        float gir = ge[j]       + s_gi[r * 768 + j];
        float giz = ge[256 + j] + s_gi[r * 768 + 256 + j];
        float gin = ge[512 + j] + s_gi[r * 768 + 512 + j];
        float ghr = s_gh[r * 768 + j]       + dec_bhh[j];
        float ghz = s_gh[r * 768 + 256 + j] + dec_bhh[256 + j];
        float ghn = s_gh[r * 768 + 512 + j] + dec_bhh[512 + j];
        float rg = 1.f / (1.f + __expf(-(gir + ghr)));
        float zg = 1.f / (1.f + __expf(-(giz + ghz)));
        float ng = tanhf(gin + rg * ghn);
        float hp = h_dec[b * 256 + j];
        float hn = (1.f - zg) * ng + zg * hp;
        h_dec[b * 256 + j] = hn;
        s_hnb[r * 256 + j] = (bf16_t)hn;
    }
    __syncthreads();

    // phase 8: pre = hn @ pWh^T + ctx @ pWc^T (WMMA)
    for (int i = 0; i < 2; ++i) {
        int nt = wave * 2 + i;
        v8f acc = {};
        acc = wmma_tile(s_hnb, 0, 256, pWh, nt * 16, 256, 256, acc);
        acc = wmma_tile(s_ctxb, 0, 512, pWc, nt * 16, 512, 512, acc);
        store_tile_f32(s_pre, 0, 256, nt * 16, acc);
    }
    __syncthreads();

    // phase 9a: add embed part of pre
    for (int i = tid; i < 16 * 256; i += 256) {
        int r = i >> 8;
        int tok = tokens[(row0 + r) * TN + t];
        s_pre[i] += preE_tab[tok * 256 + (i & 255)];
    }
    __syncthreads();

    // phase 9b: logits = pre @ gen_W^T (V=39)
    for (int i = tid; i < 16 * 39; i += 256) {
        int r = i / 39, v = i - r * 39;
        const float* p = s_pre + r * 256;
        const float* g = genW + v * 256;
        float s = 0.f;
        for (int k = 0; k < 256; ++k) s = fmaf(p[k], g[k], s);
        s_log[i] = s;
    }
    __syncthreads();

    // phase 9c: per-row NLL
    if (tid < 16) {
        int b = row0 + tid;
        const float* lg = s_log + tid * 39;
        float m = lg[0];
        for (int v = 1; v < 39; ++v) m = fmaxf(m, lg[v]);
        float sum = 0.f;
        for (int v = 0; v < 39; ++v) sum += __expf(lg[v] - m);
        float logZ = m + __logf(sum);
        int y = tokens[b * TN + t + 1];
        loss_part[t * BN + b] = (y != 0) ? (logZ - lg[y]) : 0.f;
    }
}

__global__ __launch_bounds__(256) void loss_reduce_kernel(
    const float* __restrict__ part, float* __restrict__ out) {
    __shared__ float s[256];
    int tid = threadIdx.x;
    float acc = 0.f;
    for (int i = tid; i < 127 * BN; i += 256) acc += part[i];
    s[tid] = acc;
    __syncthreads();
    for (int off = 128; off > 0; off >>= 1) {
        if (tid < off) s[tid] += s[tid + off];
        __syncthreads();
    }
    if (tid == 0) out[0] = s[0];
}

// ---------------------------------------------------------------------------
// Host launcher
// ---------------------------------------------------------------------------
extern "C" void kernel_launch(void* const* d_in, const int* in_sizes, int n_in,
                              void* d_out, int out_size, void* d_ws, size_t ws_size,
                              hipStream_t stream) {
    const int*   tokens    = (const int*)  d_in[0];
    const float* src_emb   = (const float*)d_in[1];
    const float* trg_emb   = (const float*)d_in[2];
    const float* enc_Wih_f = (const float*)d_in[3];
    const float* enc_Whh_f = (const float*)d_in[4];
    const float* enc_bih_f = (const float*)d_in[5];
    const float* enc_bhh_f = (const float*)d_in[6];
    const float* enc_Wih_b = (const float*)d_in[7];
    const float* enc_Whh_b = (const float*)d_in[8];
    const float* enc_bih_b = (const float*)d_in[9];
    const float* enc_bhh_b = (const float*)d_in[10];
    const float* bridge_W  = (const float*)d_in[11];
    const float* bridge_b  = (const float*)d_in[12];
    const float* attn_Wk   = (const float*)d_in[13];
    const float* attn_Wq   = (const float*)d_in[14];
    const float* attn_We   = (const float*)d_in[15];
    const float* dec_Wih   = (const float*)d_in[16];
    const float* dec_Whh   = (const float*)d_in[17];
    const float* dec_bih   = (const float*)d_in[18];
    const float* dec_bhh   = (const float*)d_in[19];
    const float* pre_W     = (const float*)d_in[20];
    const float* gen_W     = (const float*)d_in[21];
    float* out = (float*)d_out;

    // Workspace layout (256B aligned)
    char* ws = (char*)d_ws;
    size_t off = 0;
    auto alloc = [&](size_t bytes) {
        size_t o = off;
        off += (bytes + 255) & ~(size_t)255;
        return o;
    };
    float*  giF    = (float*)(ws + alloc((size_t)VN * 768 * 4));
    float*  giB    = (float*)(ws + alloc((size_t)VN * 768 * 4));
    float*  decE   = (float*)(ws + alloc((size_t)VN * 768 * 4));
    float*  preE   = (float*)(ws + alloc((size_t)VN * 256 * 4));
    bf16_t* WhhF   = (bf16_t*)(ws + alloc((size_t)768 * 256 * 2));
    bf16_t* WhhB   = (bf16_t*)(ws + alloc((size_t)768 * 256 * 2));
    bf16_t* dWhh   = (bf16_t*)(ws + alloc((size_t)768 * 256 * 2));
    bf16_t* dWihC  = (bf16_t*)(ws + alloc((size_t)768 * 512 * 2));
    bf16_t* WQ     = (bf16_t*)(ws + alloc((size_t)256 * 256 * 2));
    bf16_t* WK     = (bf16_t*)(ws + alloc((size_t)256 * 512 * 2));
    bf16_t* PWH    = (bf16_t*)(ws + alloc((size_t)256 * 256 * 2));
    bf16_t* PWC    = (bf16_t*)(ws + alloc((size_t)256 * 512 * 2));
    float*  hF     = (float*)(ws + alloc((size_t)BN * HN * 4));
    float*  hB     = (float*)(ws + alloc((size_t)BN * HN * 4));
    float*  hDEC   = (float*)(ws + alloc((size_t)BN * HN * 4));
    bf16_t* encBF  = (bf16_t*)(ws + alloc((size_t)BN * TN * 512 * 2));
    float*  projK  = (float*)(ws + alloc((size_t)BN * TN * 256 * 4));
    float*  lpart  = (float*)(ws + alloc((size_t)127 * BN * 4));
    (void)ws_size; (void)in_sizes; (void)n_in; (void)out_size;

    // Idempotent host-side config (same every call; not a stream op).
    hipFuncSetAttribute((const void*)dec_step_kernel,
                        hipFuncAttributeMaxDynamicSharedMemorySize, SMEM_DEC);

    // ---- setup ----
    zero_kernel<<<64, 256, 0, stream>>>(hF, BN * HN);
    zero_kernel<<<64, 256, 0, stream>>>(hB, BN * HN);
    table_kernel<<<VN, 256, 0, stream>>>(enc_Wih_f, 512, 0, enc_bih_f, src_emb, 768, giF);
    table_kernel<<<VN, 256, 0, stream>>>(enc_Wih_b, 512, 0, enc_bih_b, src_emb, 768, giB);
    table_kernel<<<VN, 256, 0, stream>>>(dec_Wih, 1024, 0, dec_bih, trg_emb, 768, decE);
    table_kernel<<<VN, 256, 0, stream>>>(pre_W, 1280, 0, nullptr, trg_emb, 256, preE);
    cvt_kernel<<<256, 256, 0, stream>>>(enc_Whh_f, 256, 0, WhhF, 768, 256);
    cvt_kernel<<<256, 256, 0, stream>>>(enc_Whh_b, 256, 0, WhhB, 768, 256);
    cvt_kernel<<<256, 256, 0, stream>>>(dec_Whh, 256, 0, dWhh, 768, 256);
    cvt_kernel<<<256, 256, 0, stream>>>(dec_Wih, 1024, 512, dWihC, 768, 512);
    cvt_kernel<<<256, 256, 0, stream>>>(attn_Wq, 256, 0, WQ, 256, 256);
    cvt_kernel<<<256, 256, 0, stream>>>(attn_Wk, 512, 0, WK, 256, 512);
    cvt_kernel<<<256, 256, 0, stream>>>(pre_W, 1280, 512, PWH, 256, 256);
    cvt_kernel<<<256, 256, 0, stream>>>(pre_W, 1280, 768, PWC, 256, 512);

    // ---- bidirectional GRU encoder ----
    for (int t = 0; t < TN; ++t)
        enc_step_kernel<<<64, 256, 0, stream>>>(t, tokens, giF, giB, WhhF, WhhB,
                                                enc_bhh_f, enc_bhh_b, hF, hB, encBF);

    // ---- bridge + attention keys ----
    bridge_kernel<<<BN, 256, 0, stream>>>(hF, hB, bridge_W, bridge_b, hDEC);
    projkey_kernel<<<1024, 256, 0, stream>>>(encBF, WK, projK);

    // ---- attention GRU decoder + per-step loss ----
    for (int t = 0; t < TN - 1; ++t)
        dec_step_kernel<<<32, 256, SMEM_DEC, stream>>>(
            t, tokens, projK, encBF, attn_We, decE, preE,
            WQ, dWhh, dWihC, PWH, PWC, dec_bhh, gen_W, hDEC, lpart);

    // ---- deterministic final reduction ----
    loss_reduce_kernel<<<1, 256, 0, stream>>>(lpart, out);
}
